// Attention_40029095198763
// MI455X (gfx1250) — matrix operand order repaired and checked by reading, not compiled, not run on hardware
//
#include <hip/hip_runtime.h>
#include <hip/hip_bf16.h>
#include <stdint.h>

// ---------------------------------------------------------------------------
// SAGAN self-attention block for MI455X (gfx1250), wave32 + WMMA bf16 path.
// Double-buffered async global->LDS staging (ASYNCcnt); 2x2 register tiling
// per wave (2 LDS b128 loads per WMMA); all fragment loads are b128.
// ---------------------------------------------------------------------------

typedef __attribute__((ext_vector_type(16))) __bf16 v16bf;
typedef __attribute__((ext_vector_type(8)))  float  v8f;

#define BATCH 8
#define CH    512
#define C8    64
#define C2    256
#define NPIX  4096   // 64*64
#define MPIX  1024   // 32*32

union frag_u {
  v16bf v;
  uint4 q[2];
};

// ----- WMMA fragment helpers (wave32, 16x16x32 bf16) ------------------------
// A-matrix 16x32 (MxK): lane l holds row M = l&15, half g = l>>4.
// Elements 0..7 hold K = g*8 + (0..7); elements 8..15 hold K = 16+g*8+(0..7).
__device__ __forceinline__ v16bf load_frag_a(const __bf16* src, int ld) {
  int lane = threadIdx.x & 31;
  int r = lane & 15, g = lane >> 4;
  const __bf16* p = src + r * ld + g * 8;
  frag_u u;
  u.q[0] = *(const uint4*)(p);
  u.q[1] = *(const uint4*)(p + 16);
  return u.v;
}

// B-matrix 32x16 (KxN) from a TRANSPOSED source ([N][K] row-major, stride ld):
// lane l holds col N = l&15, elements 0..7 hold K = g*8 + e; 8..15 hold K = 16+g*8+(e-8).
__device__ __forceinline__ v16bf load_frag_bt(const __bf16* src, int ld) {
  int lane = threadIdx.x & 31;
  int c = lane & 15, g = lane >> 4;
  const __bf16* p = src + c * ld + g * 8;
  frag_u u;
  u.q[0] = *(const uint4*)(p);
  u.q[1] = *(const uint4*)(p + 16);
  return u.v;
}

__device__ __forceinline__ v8f wmma_bf16(v16bf a, v16bf b, v8f c) {
  return __builtin_amdgcn_wmma_f32_16x16x32_bf16(false, a, false, b, (short)0, c,
                                                 false, false);
}

// ----- CDNA5 async global->LDS copy (ASYNCcnt) ------------------------------
__device__ __forceinline__ void async_copy_b128(const void* gptr, void* lptr) {
  unsigned laddr = (unsigned)(uintptr_t)lptr;  // LDS aperture: low 32b = offset
  asm volatile("global_load_async_to_lds_b128 %0, %1, off"
               :: "v"(laddr), "v"(gptr) : "memory");
}

__device__ __forceinline__ void wait_async0() {
#if __has_builtin(__builtin_amdgcn_s_wait_asynccnt)
  __builtin_amdgcn_s_wait_asynccnt(0);
#else
  asm volatile("s_wait_asynccnt 0" ::: "memory");
#endif
}

// ----- fp32 -> bf16 convert (vectorized, for weights) -----------------------
__global__ void cvt_bf16_kernel(const float4* __restrict__ src,
                                __bf16* __restrict__ dst, int n4) {
  int i = blockIdx.x * blockDim.x + threadIdx.x;
  int stride = gridDim.x * blockDim.x;
  for (; i < n4; i += stride) {
    float4 f = src[i];
    __bf16 o[4] = {(__bf16)f.x, (__bf16)f.y, (__bf16)f.z, (__bf16)f.w};
    *(uint2*)(dst + (size_t)i * 4) = *(uint2*)o;
  }
}

// ----- fp32 x[B][C][n] -> bf16 X^T[B][n][C] (tiled LDS transpose) -----------
__global__ void __launch_bounds__(256)
cvt_transpose_kernel(const float* __restrict__ x, __bf16* __restrict__ XT) {
  __shared__ float t[32][33];
  const int n0 = blockIdx.x * 32, c0 = blockIdx.y * 32, b = blockIdx.z;
  const int tid = threadIdx.x;
  const int r = tid >> 3;        // 0..31
  const int q = (tid & 7) * 4;   // 0,4,..,28
  float4 f = *(const float4*)(x + ((size_t)b * CH + c0 + r) * NPIX + n0 + q);
  t[r][q + 0] = f.x; t[r][q + 1] = f.y; t[r][q + 2] = f.z; t[r][q + 3] = f.w;
  __syncthreads();
  __bf16 o[4];
#pragma unroll
  for (int i = 0; i < 4; ++i) o[i] = (__bf16)t[q + i][r];
  *(uint2*)(XT + ((size_t)b * NPIX + n0 + r) * CH + c0 + q) = *(uint2*)o;
}

// ----- projection GEMM: D^T[4096][Co] = (W[Co][512] @ X[512][4096])^T -------
// 128 threads = 4 waves; 64x64 block tile; each wave owns a 32x32 quadrant
// (2x2 WMMA tiles -> 2 LDS b128 loads per WMMA).
__global__ void __launch_bounds__(128)
proj_gemm_kernel(const __bf16* __restrict__ W, const __bf16* __restrict__ XT,
                 __bf16* __restrict__ out, int Co) {
  const int N = NPIX, K = CH, T = K / 32;
  const int b  = blockIdx.z;
  const int n0 = blockIdx.x * 64;
  const int m0 = blockIdx.y * 64;
  const __bf16* XTb = XT + (size_t)b * N * K;

  __shared__ __align__(16) __bf16 lA[2][64 * 32];  // W tile    [64 co][32 k]
  __shared__ __align__(16) __bf16 lB[2][64 * 32];  // X^T tile  [64 n][32 k]

  const int tid  = threadIdx.x;
  const int wave = tid >> 5;
  const int lane = tid & 31;
  const int qm   = (wave >> 1) * 32;   // quadrant row offset (co)
  const int qn   = (wave & 1) * 32;    // quadrant col offset (n)

  auto issue = [&](int buf, int k0) {
    int r = tid >> 2, c = (tid & 3) * 8;  // r: 0..31
    async_copy_b128(W + (size_t)(m0 + r) * K + k0 + c, &lA[buf][r * 32 + c]);
    async_copy_b128(W + (size_t)(m0 + 32 + r) * K + k0 + c,
                    &lA[buf][(r + 32) * 32 + c]);
    async_copy_b128(XTb + (size_t)(n0 + r) * K + k0 + c, &lB[buf][r * 32 + c]);
    async_copy_b128(XTb + (size_t)(n0 + 32 + r) * K + k0 + c,
                    &lB[buf][(r + 32) * 32 + c]);
  };

  v8f acc[2][2] = {};
  issue(0, 0);
  for (int t = 0; t < T; ++t) {
    int cur = t & 1;
    wait_async0();
    __syncthreads();
    if (t + 1 < T) issue(1 - cur, (t + 1) * 32);
    v16bf a0 = load_frag_a(&lA[cur][(qm)      * 32], 32);
    v16bf a1 = load_frag_a(&lA[cur][(qm + 16) * 32], 32);
    v16bf b0 = load_frag_bt(&lB[cur][(qn)      * 32], 32);
    v16bf b1 = load_frag_bt(&lB[cur][(qn + 16) * 32], 32);
    acc[0][0] = wmma_bf16(a0, b0, acc[0][0]);
    acc[0][1] = wmma_bf16(a0, b1, acc[0][1]);
    acc[1][0] = wmma_bf16(a1, b0, acc[1][0]);
    acc[1][1] = wmma_bf16(a1, b1, acc[1][1]);
  }

  const int col = lane & 15, g = lane >> 4;
  __bf16* dst = out + (size_t)b * N * Co;  // [n][Co]
#pragma unroll
  for (int i = 0; i < 2; ++i) {
#pragma unroll
    for (int j = 0; j < 2; ++j) {
      int mm = m0 + qm + i * 16;
      int nn = n0 + qn + j * 16;
      __bf16 o[8];
#pragma unroll
      for (int v = 0; v < 8; ++v) o[v] = (__bf16)acc[i][j][v];
      *(uint4*)(dst + (size_t)(nn + col) * Co + mm + g * 8) = *(uint4*)o;
    }
  }
}

// ----- 2x2 max pool on transposed feature maps ------------------------------
// src [B][4096][C]; transposed_out=0 -> dst [B][1024][C] (for phi^T),
// transposed_out=1 -> dst [B][C][1024] (for g). 4 channels per thread.
__global__ void maxpool2_kernel(const __bf16* __restrict__ src,
                                __bf16* __restrict__ dst, int C, int total4,
                                int transposed_out) {
  int i = blockIdx.x * blockDim.x + threadIdx.x;
  if (i >= total4) return;
  int cq = i % (C / 4);
  int m  = (i / (C / 4)) % MPIX;
  int b  = i / ((C / 4) * MPIX);
  int c  = cq * 4;
  int mh = m >> 5, mw = m & 31;
  int p  = (mh * 2) * 64 + mw * 2;
  const __bf16* s = src + (size_t)b * NPIX * C + c;
  __bf16 a0[4], a1[4], a2[4], a3[4], o[4];
  *(uint2*)a0 = *(const uint2*)(s + (size_t)(p) * C);
  *(uint2*)a1 = *(const uint2*)(s + (size_t)(p + 1) * C);
  *(uint2*)a2 = *(const uint2*)(s + (size_t)(p + 64) * C);
  *(uint2*)a3 = *(const uint2*)(s + (size_t)(p + 65) * C);
#pragma unroll
  for (int q = 0; q < 4; ++q) {
    float v = fmaxf(fmaxf((float)a0[q], (float)a1[q]),
                    fmaxf((float)a2[q], (float)a3[q]));
    o[q] = (__bf16)v;
  }
  if (transposed_out) {
#pragma unroll
    for (int q = 0; q < 4; ++q)
      dst[((size_t)b * C + c + q) * MPIX + m] = o[q];
  } else {
    *(uint2*)(dst + ((size_t)b * MPIX + m) * C + c) = *(uint2*)o;
  }
}

// ----- fused attention core -------------------------------------------------
// Per (batch, 16 n-rows): S = thetaT_blk @ phi -> softmax(fp32 in LDS)
//                         attT_blk = P @ g^T   -> [16][256] bf16
__global__ void __launch_bounds__(256)
attention_core_kernel(const __bf16* __restrict__ thetaT,  // [B][4096][64]
                      const __bf16* __restrict__ phiT,    // [B][1024][64]
                      const __bf16* __restrict__ g,       // [B][256][1024]
                      __bf16* __restrict__ attT) {        // [B][4096][256]
  const int n0 = blockIdx.x * 16;
  const int b  = blockIdx.y;

  __shared__ __align__(16) float  S[16 * 1024];      // 64 KB fp32 scores
  __shared__ __align__(16) __bf16 P[16 * 1024];      // 32 KB bf16 probs
  __shared__ __align__(16) __bf16 lTh[16 * 64];      //  2 KB theta block
  __shared__ __align__(16) __bf16 lG[2][256 * 32];   // 32 KB g k-slabs

  const int tid = threadIdx.x, wave = tid >> 5, lane = tid & 31;

  {  // stage theta block [16 n][64 c]
    int idx = tid * 4;
    int r = idx >> 6, c = idx & 63;
    *(uint2*)&lTh[idx] =
        *(const uint2*)(thetaT + ((size_t)b * NPIX + n0 + r) * C8 + c);
  }
  __syncthreads();

  // ---- scores: 64 m-tiles, K=64 (2 wmma each); 8 m-tiles per wave ----
  const __bf16* phiTb = phiT + (size_t)b * MPIX * C8;
  v16bf a0 = load_frag_a(&lTh[0], 64);   // k = 0..31
  v16bf a1 = load_frag_a(&lTh[32], 64);  // k = 32..63
#pragma unroll 2
  for (int t = 0; t < 8; ++t) {
    int mb = (wave * 8 + t) * 16;
    v8f acc = {};
    v16bf b0 = load_frag_bt(phiTb + (size_t)mb * C8, C8);       // c = 0..31
    v16bf b1 = load_frag_bt(phiTb + (size_t)mb * C8 + 32, C8);  // c = 32..63
    acc = wmma_bf16(a0, b0, acc);
    acc = wmma_bf16(a1, b1, acc);
    int col = lane & 15, gg = lane >> 4;
#pragma unroll
    for (int v = 0; v < 8; ++v)
      S[(gg * 8 + v) * 1024 + mb + col] = acc[v];
  }
  __syncthreads();

  // ---- softmax over m (1024) for each of 16 rows; 2 rows / wave ----
#pragma unroll
  for (int rr = 0; rr < 2; ++rr) {
    int row = wave * 2 + rr;
    float* Sr = &S[row * 1024];
    float mx = -3.0e38f;
    for (int i = lane; i < 1024; i += 32) mx = fmaxf(mx, Sr[i]);
#pragma unroll
    for (int off = 16; off > 0; off >>= 1) mx = fmaxf(mx, __shfl_xor(mx, off, 32));
    float sum = 0.f;
    for (int i = lane; i < 1024; i += 32) {
      float e = __expf(Sr[i] - mx);
      Sr[i] = e;
      sum += e;
    }
#pragma unroll
    for (int off = 16; off > 0; off >>= 1) sum += __shfl_xor(sum, off, 32);
    float inv = 1.0f / sum;
    for (int i = lane; i < 1024; i += 32) P[row * 1024 + i] = (__bf16)(Sr[i] * inv);
  }
  __syncthreads();

  // ---- attend: attT_blk[16][256] = P[16][1024] @ g^T; 2 c2-tiles / wave ----
  const __bf16* gb = g + (size_t)b * C2 * MPIX;
  const int c0 = wave * 32;

  auto issueG = [&](int buf, int k0) {
    const __bf16* src = gb + (size_t)tid * MPIX + k0;
#pragma unroll
    for (int q = 0; q < 4; ++q)
      async_copy_b128(src + q * 8, &lG[buf][tid * 32 + q * 8]);
  };

  v8f acc0 = {}, acc1 = {};
  issueG(0, 0);
  for (int t = 0; t < 32; ++t) {
    int cur = t & 1;
    wait_async0();
    __syncthreads();
    if (t + 1 < 32) issueG(1 - cur, (t + 1) * 32);
    v16bf a  = load_frag_a(&P[t * 32], 1024);
    v16bf b0 = load_frag_bt(&lG[cur][c0 * 32], 32);
    v16bf b1 = load_frag_bt(&lG[cur][(c0 + 16) * 32], 32);
    acc0 = wmma_bf16(a, b0, acc0);
    acc1 = wmma_bf16(a, b1, acc1);
  }

  __bf16* dst = attT + ((size_t)b * NPIX + n0) * C2;
  const int col = lane & 15, gg = lane >> 4;
#pragma unroll
  for (int v = 0; v < 8; ++v) {
    dst[(size_t)(gg * 8 + v) * C2 + c0 + col]      = (__bf16)acc0[v];
    dst[(size_t)(gg * 8 + v) * C2 + c0 + 16 + col] = (__bf16)acc1[v];
  }
}

// ----- output projection + residual:  out = gamma * (Wo @ att) + x ----------
// 128 threads = 4 waves; 64x64 block tile; 2x2 WMMA tiles per wave.
__global__ void __launch_bounds__(128)
output_gemm_kernel(const __bf16* __restrict__ Wo,    // [512][256]
                   const __bf16* __restrict__ attT,  // [B][4096][256]
                   const float* __restrict__ x,      // [B][512][4096]
                   const float* __restrict__ gamma_p,
                   float* __restrict__ out) {
  const int N = NPIX, K = C2, Co = CH, T = K / 32;
  const int b  = blockIdx.z;
  const int n0 = blockIdx.x * 64;
  const int m0 = blockIdx.y * 64;
  const float gamma = *gamma_p;

  __shared__ __align__(16) __bf16 lA[2][64 * 32];  // Wo tile    [64 oc][32 k]
  __shared__ __align__(16) __bf16 lB[2][64 * 32];  // attT tile  [64 p][32 k]

  const int tid = threadIdx.x, wave = tid >> 5, lane = tid & 31;
  const int qm = (wave >> 1) * 32;   // quadrant row offset (oc)
  const int qn = (wave & 1) * 32;    // quadrant col offset (p)
  const __bf16* at = attT + ((size_t)b * NPIX + n0) * K;

  auto issue = [&](int buf, int k0) {
    int r = tid >> 2, c = (tid & 3) * 8;  // r: 0..31
    async_copy_b128(Wo + (size_t)(m0 + r) * K + k0 + c, &lA[buf][r * 32 + c]);
    async_copy_b128(Wo + (size_t)(m0 + 32 + r) * K + k0 + c,
                    &lA[buf][(r + 32) * 32 + c]);
    async_copy_b128(at + (size_t)r * K + k0 + c, &lB[buf][r * 32 + c]);
    async_copy_b128(at + (size_t)(r + 32) * K + k0 + c,
                    &lB[buf][(r + 32) * 32 + c]);
  };

  v8f acc[2][2] = {};
  issue(0, 0);
  for (int t = 0; t < T; ++t) {
    int cur = t & 1;
    wait_async0();
    __syncthreads();
    if (t + 1 < T) issue(1 - cur, (t + 1) * 32);
    v16bf a0 = load_frag_a(&lA[cur][(qm)      * 32], 32);
    v16bf a1 = load_frag_a(&lA[cur][(qm + 16) * 32], 32);
    v16bf b0 = load_frag_bt(&lB[cur][(qn)      * 32], 32);
    v16bf b1 = load_frag_bt(&lB[cur][(qn + 16) * 32], 32);
    acc[0][0] = wmma_bf16(a0, b0, acc[0][0]);
    acc[0][1] = wmma_bf16(a0, b1, acc[0][1]);
    acc[1][0] = wmma_bf16(a1, b0, acc[1][0]);
    acc[1][1] = wmma_bf16(a1, b1, acc[1][1]);
  }

  const int col = lane & 15, g = lane >> 4;
  const float* xb = x + (size_t)b * Co * N;
  float* ob = out + (size_t)b * Co * N;
#pragma unroll
  for (int i = 0; i < 2; ++i) {
#pragma unroll
    for (int j = 0; j < 2; ++j) {
#pragma unroll
      for (int v = 0; v < 8; ++v) {
        int row = m0 + qm + i * 16 + g * 8 + v;
        int cc  = n0 + qn + j * 16 + col;
        ob[(size_t)row * N + cc] =
            gamma * acc[i][j][v] + xb[(size_t)row * N + cc];
      }
    }
  }
}

// ---------------------------------------------------------------------------
extern "C" void kernel_launch(void* const* d_in, const int* in_sizes, int n_in,
                              void* d_out, int out_size, void* d_ws, size_t ws_size,
                              hipStream_t stream) {
  const float* x       = (const float*)d_in[0];
  const float* w_theta = (const float*)d_in[1];
  const float* w_phi   = (const float*)d_in[2];
  const float* w_g     = (const float*)d_in[3];
  const float* w_o     = (const float*)d_in[4];
  const float* gamma   = (const float*)d_in[5];
  float* out = (float*)d_out;

  char* ws = (char*)d_ws;
  size_t off = 0;
  auto alloc = [&](size_t elems) -> __bf16* {
    __bf16* p = (__bf16*)(ws + off);
    off += (elems * sizeof(__bf16) + 255) & ~(size_t)255;
    return p;
  };

  __bf16* XT     = alloc((size_t)BATCH * NPIX * CH);   // x^T  [B][n][C]
  __bf16* Wt     = alloc((size_t)C8 * CH);
  __bf16* Wp     = alloc((size_t)C8 * CH);
  __bf16* Wg     = alloc((size_t)C2 * CH);
  __bf16* Wo     = alloc((size_t)CH * C2);
  __bf16* thetaT = alloc((size_t)BATCH * NPIX * C8);   // [B][n][64]
  __bf16* phiFT  = alloc((size_t)BATCH * NPIX * C8);   // [B][n][64]
  __bf16* gFT    = alloc((size_t)BATCH * NPIX * C2);   // [B][n][256]
  __bf16* phiT   = alloc((size_t)BATCH * MPIX * C8);   // [B][m][64]
  __bf16* gP     = alloc((size_t)BATCH * C2 * MPIX);   // [B][256][m]
  __bf16* attT   = alloc((size_t)BATCH * NPIX * C2);   // [B][n][256]

  // 1) conversions: x -> X^T (tiled transpose), weights (straight)
  cvt_transpose_kernel<<<dim3(NPIX / 32, CH / 32, BATCH), 256, 0, stream>>>(x, XT);
  cvt_bf16_kernel<<<32, 256, 0, stream>>>((const float4*)w_theta, Wt, C8 * CH / 4);
  cvt_bf16_kernel<<<32, 256, 0, stream>>>((const float4*)w_phi, Wp, C8 * CH / 4);
  cvt_bf16_kernel<<<128, 256, 0, stream>>>((const float4*)w_g, Wg, C2 * CH / 4);
  cvt_bf16_kernel<<<128, 256, 0, stream>>>((const float4*)w_o, Wo, CH * C2 / 4);

  // 2) projection GEMMs (WMMA bf16, async pipelined, transposed outputs)
  proj_gemm_kernel<<<dim3(NPIX / 64, C8 / 64, BATCH), 128, 0, stream>>>(
      Wt, XT, thetaT, C8);
  proj_gemm_kernel<<<dim3(NPIX / 64, C8 / 64, BATCH), 128, 0, stream>>>(
      Wp, XT, phiFT, C8);
  proj_gemm_kernel<<<dim3(NPIX / 64, C2 / 64, BATCH), 128, 0, stream>>>(
      Wg, XT, gFT, C2);

  // 3) 2x2 max pools
  {
    int tot_phi4 = BATCH * MPIX * (C8 / 4);   // 131072
    int tot_g4   = BATCH * MPIX * (C2 / 4);   // 524288
    maxpool2_kernel<<<(tot_phi4 + 255) / 256, 256, 0, stream>>>(
        phiFT, phiT, C8, tot_phi4, /*transposed_out=*/0);
    maxpool2_kernel<<<(tot_g4 + 255) / 256, 256, 0, stream>>>(
        gFT, gP, C2, tot_g4, /*transposed_out=*/1);
  }

  // 4) fused scores -> softmax -> attend
  attention_core_kernel<<<dim3(NPIX / 16, BATCH), 256, 0, stream>>>(
      thetaT, phiT, gP, attT);

  // 5) output projection + residual
  output_gemm_kernel<<<dim3(NPIX / 64, CH / 64, BATCH), 128, 0, stream>>>(
      Wo, attT, x, gamma, out);
}
